// DemoTransformer_6674379178761
// MI455X (gfx1250) — matrix operand order repaired and checked by reading, not compiled, not run on hardware
//
#include <hip/hip_runtime.h>

#define D_MODEL 768
#define N_HEADS 12
#define D_HEAD  64
#define D_FF    3072
#define N_LAYERS 12
#define N_VOCAB 50257
#define SEQ     1024
#define BATCH   2
#define NT      (BATCH*SEQ)   // 2048 token rows

typedef __attribute__((ext_vector_type(16))) __bf16 v16bf;
typedef __attribute__((ext_vector_type(8)))  float  v8f;
typedef unsigned int u32x4 __attribute__((ext_vector_type(4)));
typedef int          i32x4 __attribute__((ext_vector_type(4)));
typedef int          i32x8 __attribute__((ext_vector_type(8)));
typedef float        f32x4 __attribute__((ext_vector_type(4)));

union Frag { v16bf v; unsigned short u[16]; u32x4 q[2]; };

__device__ inline unsigned short f32_to_bf16(float f) {
    unsigned int u = __float_as_uint(f);
    u += 0x7FFFu + ((u >> 16) & 1u);          // round-to-nearest-even
    return (unsigned short)(u >> 16);
}

__device__ inline float gelu_new(float x) {
    const float c = 0.7978845608028654f;       // sqrt(2/pi)
    float x3 = x * x * x;
    return 0.5f * x * (1.0f + tanhf(c * (x + 0.044715f * x3)));
}

// -------------------------------------------------------------------------
// Weight staging: fp32 [K,N] (optionally head-blocked) -> bf16 pre-swizzled
// into the WMMA B-fragment layout:
//   dst[((nt*KT + kt)*32 + lane)*16 + i] = bf16(W[k][n])
//   k = kt*32 + (lane>>4)*16 + i,  n = nt*16 + (lane&15)  (n clamped to N-1)
// One thread produces one lane-fragment (16 contiguous ushorts = 32B store).
// -------------------------------------------------------------------------
__global__ __launch_bounds__(256) void cvt_swizzle_bf16(
    const float* __restrict__ src, unsigned short* __restrict__ dst,
    int N, int K, int ldb, int head_ncols, long long head_stride, int ntiles)
{
    int t = blockIdx.x * 256 + threadIdx.x;
    int KT = K >> 5;
    int total = ntiles * KT * 32;
    if (t >= total) return;
    int lane = t & 31;
    int kt   = (t >> 5) % KT;
    int nt   = (t >> 5) / KT;
    int half = lane >> 4, l15 = lane & 15;
    int n = nt * 16 + l15;
    if (n >= N) n = N - 1;                  // pad tiles duplicate last col
    size_t base = (size_t)(n / head_ncols) * (size_t)head_stride
                + (size_t)(n % head_ncols);
    int k0 = kt * 32 + half * 16;
    unsigned short* d = dst + (size_t)t * 16;
#pragma unroll
    for (int i = 0; i < 16; ++i)
        d[i] = f32_to_bf16(src[base + (size_t)(k0 + i) * (size_t)ldb]);
}

// -------------------------------------------------------------------------
// Embedding: resid[b,s,:] = W_E[token] + W_pos[s]
// -------------------------------------------------------------------------
__global__ __launch_bounds__(256) void embed_kernel(
    const int* __restrict__ tok, const float* __restrict__ WE,
    const float* __restrict__ Wp, float* __restrict__ resid)
{
    int i = blockIdx.x * 256 + threadIdx.x;
    if (i >= NT * D_MODEL) return;
    int d = i % D_MODEL;
    int t = i / D_MODEL;
    int s = t % SEQ;
    int token = tok[t];
    resid[i] = WE[(size_t)token * D_MODEL + d] + Wp[(size_t)s * D_MODEL + d];
}

// -------------------------------------------------------------------------
// LayerNorm (one block per row of 768) -> bf16 output
// -------------------------------------------------------------------------
__global__ __launch_bounds__(256) void layernorm_bf16(
    const float* __restrict__ x, const float* __restrict__ w,
    const float* __restrict__ bsh, unsigned short* __restrict__ out)
{
    const int row = blockIdx.x;
    const int tid = threadIdx.x;
    const float* xr = x + (size_t)row * D_MODEL;
    float v0 = xr[tid], v1 = xr[tid + 256], v2 = xr[tid + 512];

    __shared__ float red[256];
    red[tid] = v0 + v1 + v2;
    __syncthreads();
    for (int s2 = 128; s2 > 0; s2 >>= 1) {
        if (tid < s2) red[tid] += red[tid + s2];
        __syncthreads();
    }
    float mean = red[0] * (1.0f / D_MODEL);
    __syncthreads();

    float c0 = v0 - mean, c1 = v1 - mean, c2 = v2 - mean;
    red[tid] = c0 * c0 + c1 * c1 + c2 * c2;
    __syncthreads();
    for (int s2 = 128; s2 > 0; s2 >>= 1) {
        if (tid < s2) red[tid] += red[tid + s2];
        __syncthreads();
    }
    float inv = rsqrtf(red[0] * (1.0f / D_MODEL) + 1e-5f);

    unsigned short* orow = out + (size_t)row * D_MODEL;
    orow[tid]       = f32_to_bf16(c0 * inv * w[tid]       + bsh[tid]);
    orow[tid + 256] = f32_to_bf16(c1 * inv * w[tid + 256] + bsh[tid + 256]);
    orow[tid + 512] = f32_to_bf16(c2 * inv * w[tid + 512] + bsh[tid + 512]);
}

// -------------------------------------------------------------------------
// WMMA GEMM: out[2048, N] = A_bf16[2048, K] * B[K, N] (+bias, +res, gelu, ...)
// b_bf16 != 0: Bh is pre-swizzled bf16 fragments (two b128 loads per panel).
// b_bf16 == 0: fallback, Bf is fp32 row-major/head-blocked, converted in-loop.
// All hot-loop addresses are recomputed from kernel args each iteration so
// address-space inference keeps them global_load_b128 (no flat, no DScnt).
// Workgroup 256 = 8 waves = (2 row-groups x 4 col-groups); each wave does a
// 64x32 tile = 4 A-frags x 2 B-panels = 8 WMMAs per 32-deep K step.
// -------------------------------------------------------------------------
__global__ __launch_bounds__(256) void gemm_bf16_wmma(
    const unsigned short* __restrict__ A,   // [2048, K] bf16 bits
    const unsigned short* __restrict__ Bh,  // swizzled bf16 weights (or null)
    const float* __restrict__ Bf,           // fp32 weights (or null)
    int b_bf16,
    const float* __restrict__ bias,         // [N] or nullptr
    const float* __restrict__ res,          // [2048, N] or nullptr
    void* __restrict__ out,                 // fp32 or bf16 [2048, N]
    int N, int K, int ldb,
    int head_ncols, long long head_stride,
    int do_gelu, int out_bf16)
{
    const int tid  = threadIdx.x;
    const int wave = tid >> 5;
    const int lane = tid & 31;
    const int half = lane >> 4;   // 0: lanes 0-15, 1: lanes 16-31
    const int l15  = lane & 15;
    const int mw   = wave >> 2;   // 0..1
    const int nw   = wave & 3;    // 0..3

    const int m0 = blockIdx.x * 128 + mw * 64;
    const int n0 = blockIdx.y * 128 + nw * 32;

    int col[2], colc[2];
#pragma unroll
    for (int p = 0; p < 2; ++p) {
        col[p]  = n0 + p * 16 + l15;
        colc[p] = (col[p] < N) ? col[p] : (N - 1);
    }

    const int kbA = half * 16;   // A frag: lanes 0-15 K 0-15, lanes 16-31 K 16-31
    const int kbB = half * 16;   // B frag: lanes 0-15 K 0-15, lanes 16-31 K 16-31

    v8f acc[4][2];
    v8f vz = {0.f, 0.f, 0.f, 0.f, 0.f, 0.f, 0.f, 0.f};
#pragma unroll
    for (int t = 0; t < 4; ++t)
#pragma unroll
        for (int p = 0; p < 2; ++p) acc[t][p] = vz;

    if (b_bf16) {
        // swizzled layout: tile (nt,kt): 32 lanes x 16 ushorts contiguous
        const size_t KT = (size_t)(K >> 5);
        size_t pbase[2];
#pragma unroll
        for (int p = 0; p < 2; ++p)
            pbase[p] = (size_t)(n0 / 16 + p) * KT * 512 + (size_t)lane * 16;

        for (int kk = 0; kk < K; kk += 32) {
            Frag bf[2];
#pragma unroll
            for (int p = 0; p < 2; ++p) {
                // per-32K step = 512 ushorts = kk*16; recompute from arg each
                // iteration so it stays a global (not flat) access.
                const u32x4* bp =
                    (const u32x4*)(Bh + pbase[p] + (size_t)kk * 16);
                bf[p].q[0] = bp[0];
                bf[p].q[1] = bp[1];
            }
            Frag af[4];
#pragma unroll
            for (int t = 0; t < 4; ++t) {
                const u32x4* ap = (const u32x4*)
                    (A + (size_t)(m0 + t * 16 + l15) * (size_t)K + kk + kbA);
                af[t].q[0] = ap[0];
                af[t].q[1] = ap[1];
            }
#pragma unroll
            for (int t = 0; t < 4; ++t)
#pragma unroll
                for (int p = 0; p < 2; ++p)
                    acc[t][p] = __builtin_amdgcn_wmma_f32_16x16x32_bf16(
                        false, af[t].v, false, bf[p].v, (short)0,
                        acc[t][p], false, false);
        }
    } else {
        size_t boff[2];
#pragma unroll
        for (int p = 0; p < 2; ++p)
            boff[p] = (size_t)(colc[p] / head_ncols) * (size_t)head_stride
                    + (size_t)(colc[p] % head_ncols);
        for (int kk = 0; kk < K; kk += 32) {
            Frag bf[2];
#pragma unroll
            for (int p = 0; p < 2; ++p)
#pragma unroll
                for (int i = 0; i < 16; ++i)
                    bf[p].u[i] = f32_to_bf16(
                        Bf[boff[p] + (size_t)(kk + kbB + i) * (size_t)ldb]);
            Frag af[4];
#pragma unroll
            for (int t = 0; t < 4; ++t) {
                const u32x4* ap = (const u32x4*)
                    (A + (size_t)(m0 + t * 16 + l15) * (size_t)K + kk + kbA);
                af[t].q[0] = ap[0];
                af[t].q[1] = ap[1];
            }
#pragma unroll
            for (int t = 0; t < 4; ++t)
#pragma unroll
                for (int p = 0; p < 2; ++p)
                    acc[t][p] = __builtin_amdgcn_wmma_f32_16x16x32_bf16(
                        false, af[t].v, false, bf[p].v, (short)0,
                        acc[t][p], false, false);
        }
    }

    float*          Cf = (float*)out;
    unsigned short* Ch = (unsigned short*)out;
#pragma unroll
    for (int p = 0; p < 2; ++p) {
        float bv = bias ? bias[colc[p]] : 0.0f;
        if (col[p] < N) {
#pragma unroll
            for (int t = 0; t < 4; ++t) {
#pragma unroll
                for (int r = 0; r < 8; ++r) {
                    int row = m0 + t * 16 + half * 8 + r;   // C layout per ISA
                    size_t idx = (size_t)row * (size_t)N + col[p];
                    float v = acc[t][p][r] + bv;
                    if (res)     v += res[idx];
                    if (do_gelu) v  = gelu_new(v);
                    if (out_bf16) Ch[idx] = f32_to_bf16(v);
                    else          Cf[idx] = v;
                }
            }
        }
    }
}

// -------------------------------------------------------------------------
// TDM descriptor path (gated; falls back to cooperative VMEM loads)
// -------------------------------------------------------------------------
#if defined(__gfx1250__) && \
    __has_builtin(__builtin_amdgcn_tensor_load_to_lds) && \
    __has_builtin(__builtin_amdgcn_s_wait_tensorcnt)
#define USE_TDM 1
#else
#define USE_TDM 0
#endif

#if USE_TDM
// 2D tile load: 64 rows x 64 f32, row stride 768 f32, into LDS at lds_off.
__device__ inline void tdm_load_tile_64x64_f32(unsigned lds_off,
                                               const float* gsrc)
{
    unsigned long long ga = (unsigned long long)(size_t)gsrc;
    u32x4 g0;
    g0[0] = 1u;                                           // count=1 (valid)
    g0[1] = lds_off;                                      // lds_addr (bytes)
    g0[2] = (unsigned)(ga & 0xFFFFFFFFu);                 // global_addr[31:0]
    g0[3] = (unsigned)((ga >> 32) & 0x01FFFFFFu)          // global_addr[56:32]
          | (2u << 30);                                   // type = 2 (image)
    i32x8 g1;
    g1[0] = (2 << 16);          // data_size = 2 -> 4 bytes
    g1[1] = (64 << 16);         // tensor_dim0[15:0] = 64
    g1[2] = (64 << 16);         // tensor_dim0 hi = 0 | tensor_dim1[15:0] = 64
    g1[3] = (64 << 16);         // tensor_dim1 hi = 0 | tile_dim0 = 64
    g1[4] = 64;                 // tile_dim1 = 64, tile_dim2 = 0
    g1[5] = D_MODEL;            // tensor_dim0_stride = 768 (data_size units)
    g1[6] = 0;                  // stride hi / tensor_dim1_stride = 0 (2D)
    g1[7] = 0;
    i32x4 gz4 = {0, 0, 0, 0};
    i32x8 gz8 = {0, 0, 0, 0, 0, 0, 0, 0};
    // clang-23 / amdgpu-toolchain form: 6 args (g0, g1, g2, g3, g4, cpol)
    __builtin_amdgcn_tensor_load_to_lds(g0, g1, gz4, gz4, gz8, 0);
}
#endif

// -------------------------------------------------------------------------
// Causal flash attention, one thread per query row; K/V tiles staged in LDS
// (TDM DMA when available). grid = (SEQ/128, H, B), block = 128.
// -------------------------------------------------------------------------
__global__ __launch_bounds__(128) void attn_flash(
    const float* __restrict__ qf, const float* __restrict__ kf,
    const float* __restrict__ vf, unsigned short* __restrict__ zb)
{
    const int tid = threadIdx.x;
    const int q0  = blockIdx.x * 128;
    const int h   = blockIdx.y;
    const int b   = blockIdx.z;
    const int q   = q0 + tid;

    __shared__ __align__(16) float smem[2 * 64 * 64];   // Ks | Vs
    float* Ks = smem;
    float* Vs = smem + 4096;

    const float* qrow = qf + ((size_t)(b * SEQ + q)) * D_MODEL + h * D_HEAD;
    f32x4 qv[16], acc[16];
    f32x4 z4 = {0.f, 0.f, 0.f, 0.f};
#pragma unroll
    for (int i = 0; i < 16; ++i) {
        qv[i]  = ((const f32x4*)qrow)[i];
        acc[i] = z4;
    }

    float mmax = -1e30f, lsum = 0.f;
    const int qmax = q0 + 127;

    for (int kb = 0; kb <= qmax; kb += 64) {
        __syncthreads();
#if USE_TDM
        if (tid < 32) {   // wave 0 issues the DMA; EXEC is ignored by TDM
            const float* kst = kf + ((size_t)(b * SEQ + kb)) * D_MODEL + h * D_HEAD;
            const float* vst = vf + ((size_t)(b * SEQ + kb)) * D_MODEL + h * D_HEAD;
            tdm_load_tile_64x64_f32(0u, kst);
            tdm_load_tile_64x64_f32(4096u * 4u, vst);
            __builtin_amdgcn_s_wait_tensorcnt(0);
        }
        __syncthreads();
#else
#pragma unroll
        for (int it = 0; it < 8; ++it) {
            int idx = tid + it * 128;
            int row = idx >> 4;
            int c4  = idx & 15;
            size_t g = ((size_t)(b * SEQ + kb + row)) * D_MODEL
                     + h * D_HEAD + c4 * 4;
            ((f32x4*)Ks)[idx] = *(const f32x4*)(kf + g);
            ((f32x4*)Vs)[idx] = *(const f32x4*)(vf + g);
        }
        __syncthreads();
#endif

        int jend = q - kb + 1;
        if (jend > 64) jend = 64;
        for (int j = 0; j < jend; ++j) {
            const f32x4* kr = (const f32x4*)&Ks[j * 64];
            float s = 0.f;
#pragma unroll
            for (int i = 0; i < 16; ++i) {
                f32x4 kv = kr[i];
                s += qv[i].x * kv.x + qv[i].y * kv.y
                   + qv[i].z * kv.z + qv[i].w * kv.w;
            }
            s *= 0.125f;   // 1/sqrt(64)
            float nm   = fmaxf(mmax, s);
            float corr = __expf(mmax - nm);
            float p    = __expf(s - nm);
            lsum = lsum * corr + p;
            const f32x4* vr = (const f32x4*)&Vs[j * 64];
#pragma unroll
            for (int i = 0; i < 16; ++i) {
                f32x4 vv = vr[i];
                acc[i].x = acc[i].x * corr + p * vv.x;
                acc[i].y = acc[i].y * corr + p * vv.y;
                acc[i].z = acc[i].z * corr + p * vv.z;
                acc[i].w = acc[i].w * corr + p * vv.w;
            }
            mmax = nm;
        }
    }

    float inv = 1.f / lsum;
    unsigned short* dst = zb + ((size_t)(b * SEQ + q)) * D_MODEL + h * D_HEAD;
#pragma unroll
    for (int i = 0; i < 16; ++i) {
        dst[i * 4 + 0] = f32_to_bf16(acc[i].x * inv);
        dst[i * 4 + 1] = f32_to_bf16(acc[i].y * inv);
        dst[i * 4 + 2] = f32_to_bf16(acc[i].z * inv);
        dst[i * 4 + 3] = f32_to_bf16(acc[i].w * inv);
    }
}

// -------------------------------------------------------------------------
extern "C" void kernel_launch(void* const* d_in, const int* in_sizes, int n_in,
                              void* d_out, int out_size, void* d_ws, size_t ws_size,
                              hipStream_t stream) {
    (void)in_sizes; (void)n_in; (void)out_size;

    const int*   tokens = (const int*)  d_in[0];
    const float* W_E    = (const float*)d_in[1];
    const float* W_pos  = (const float*)d_in[2];
    const float* ln1w   = (const float*)d_in[3];
    const float* ln1b   = (const float*)d_in[4];
    const float* W_Q    = (const float*)d_in[5];
    const float* b_Q    = (const float*)d_in[6];
    const float* W_K    = (const float*)d_in[7];
    const float* b_K    = (const float*)d_in[8];
    const float* W_V    = (const float*)d_in[9];
    const float* b_V    = (const float*)d_in[10];
    const float* W_O    = (const float*)d_in[11];
    const float* b_O    = (const float*)d_in[12];
    const float* ln2w   = (const float*)d_in[13];
    const float* ln2b   = (const float*)d_in[14];
    const float* W_in   = (const float*)d_in[15];
    const float* b_in   = (const float*)d_in[16];
    const float* W_out  = (const float*)d_in[17];
    const float* b_out  = (const float*)d_in[18];
    const float* lnfw   = (const float*)d_in[19];
    const float* lnfb   = (const float*)d_in[20];
    const float* W_U    = (const float*)d_in[21];
    const float* b_U    = (const float*)d_in[22];

    // ---- workspace carve --------------------------------------------------
    const size_t TD = (size_t)NT * D_MODEL;            // 1.57M elements
    float* resid = (float*)d_ws;
    float* qf    = resid + TD;
    float* kf    = qf + TD;
    float* vf    = kf + TD;
    unsigned short* xb  = (unsigned short*)(vf + TD);
    unsigned short* zb  = xb + TD;
    unsigned short* h1b = zb + TD;                     // NT * D_FF bf16
    size_t actBytes = 4 * TD * 4 + 2 * TD * 2 + (size_t)NT * D_FF * 2; // ~44MB

    // staging: padded tile count covers the full GEMM grid for W_U
    const int vtiles = ((N_VOCAB + 127) / 128) * 8;    // 3144 tiles of 16 cols
    const size_t maxSlice = (size_t)vtiles * 16 * D_MODEL;  // 38.6M elems
    unsigned short* wstage = (unsigned short*)((char*)d_ws + actBytes);
    const int staged = (ws_size >= actBytes + maxSlice * 2) ? 1 : 0;

    const int PLAIN = 1 << 30;    // head_ncols for plain row-major weights

    auto stage = [&](const float* W, int N, int K, int ldb,
                     int head_ncols, long long head_stride) {
        if (!staged) return;
        int ntiles = ((N + 127) / 128) * 8;
        size_t total = (size_t)ntiles * (size_t)(K >> 5) * 32;
        cvt_swizzle_bf16<<<dim3((unsigned)((total + 255) / 256)), dim3(256),
                           0, stream>>>(
            W, wstage, N, K, ldb, head_ncols, head_stride, ntiles);
    };
    const unsigned short* BH = staged ? wstage : nullptr;

    embed_kernel<<<dim3((NT * D_MODEL + 255) / 256), dim3(256), 0, stream>>>(
        tokens, W_E, W_pos, resid);

    dim3 blk(256);
    dim3 gD (NT / 128, D_MODEL / 128);           // (16, 6)
    dim3 gFF(NT / 128, D_FF / 128);              // (16, 24)
    dim3 gV (NT / 128, (N_VOCAB + 127) / 128);   // (16, 393)

    const size_t wqkv = (size_t)N_HEADS * D_MODEL * D_HEAD;   // per-layer slice
    const long long hstr = (long long)D_MODEL * D_HEAD;       // 49152

    for (int l = 0; l < N_LAYERS; ++l) {
        layernorm_bf16<<<NT, 256, 0, stream>>>(
            resid, ln1w + (size_t)l * D_MODEL, ln1b + (size_t)l * D_MODEL, xb);

        stage(W_Q + (size_t)l * wqkv, D_MODEL, D_MODEL, D_HEAD, D_HEAD, hstr);
        gemm_bf16_wmma<<<gD, blk, 0, stream>>>(
            xb, BH, W_Q + (size_t)l * wqkv, staged,
            b_Q + (size_t)l * N_HEADS * D_HEAD,
            nullptr, qf, D_MODEL, D_MODEL, D_HEAD, D_HEAD, hstr, 0, 0);

        stage(W_K + (size_t)l * wqkv, D_MODEL, D_MODEL, D_HEAD, D_HEAD, hstr);
        gemm_bf16_wmma<<<gD, blk, 0, stream>>>(
            xb, BH, W_K + (size_t)l * wqkv, staged,
            b_K + (size_t)l * N_HEADS * D_HEAD,
            nullptr, kf, D_MODEL, D_MODEL, D_HEAD, D_HEAD, hstr, 0, 0);

        stage(W_V + (size_t)l * wqkv, D_MODEL, D_MODEL, D_HEAD, D_HEAD, hstr);
        gemm_bf16_wmma<<<gD, blk, 0, stream>>>(
            xb, BH, W_V + (size_t)l * wqkv, staged,
            b_V + (size_t)l * N_HEADS * D_HEAD,
            nullptr, vf, D_MODEL, D_MODEL, D_HEAD, D_HEAD, hstr, 0, 0);

        attn_flash<<<dim3(SEQ / 128, N_HEADS, BATCH), dim3(128), 0, stream>>>(
            qf, kf, vf, zb);

        // attn out projection, accumulate into resid (W_O flat = [768,768] RM)
        stage(W_O + (size_t)l * D_MODEL * D_MODEL,
              D_MODEL, D_MODEL, D_MODEL, PLAIN, 0);
        gemm_bf16_wmma<<<gD, blk, 0, stream>>>(
            zb, BH, W_O + (size_t)l * D_MODEL * D_MODEL, staged,
            b_O + (size_t)l * D_MODEL,
            resid, resid, D_MODEL, D_MODEL, D_MODEL, PLAIN, 0, 0, 0);

        layernorm_bf16<<<NT, 256, 0, stream>>>(
            resid, ln2w + (size_t)l * D_MODEL, ln2b + (size_t)l * D_MODEL, xb);

        // MLP in: gelu, bf16 output
        stage(W_in + (size_t)l * D_MODEL * D_FF,
              D_FF, D_MODEL, D_FF, PLAIN, 0);
        gemm_bf16_wmma<<<gFF, blk, 0, stream>>>(
            xb, BH, W_in + (size_t)l * D_MODEL * D_FF, staged,
            b_in + (size_t)l * D_FF,
            nullptr, h1b, D_FF, D_MODEL, D_FF, PLAIN, 0, 1, 1);

        // MLP out: accumulate into resid
        stage(W_out + (size_t)l * D_FF * D_MODEL,
              D_MODEL, D_FF, D_MODEL, PLAIN, 0);
        gemm_bf16_wmma<<<gD, blk, 0, stream>>>(
            h1b, BH, W_out + (size_t)l * D_FF * D_MODEL, staged,
            b_out + (size_t)l * D_MODEL,
            resid, resid, D_MODEL, D_FF, D_MODEL, PLAIN, 0, 0, 0);
    }

    layernorm_bf16<<<NT, 256, 0, stream>>>(resid, lnfw, lnfb, xb);

    // Unembedding -> fp32 logits in d_out
    stage(W_U, N_VOCAB, D_MODEL, N_VOCAB, PLAIN, 0);
    gemm_bf16_wmma<<<gV, blk, 0, stream>>>(
        xb, BH, W_U, staged, b_U, nullptr, d_out,
        N_VOCAB, D_MODEL, N_VOCAB, PLAIN, 0, 0, 0);
}
